// VQVAE_69896297775278
// MI455X (gfx1250) — matrix operand order repaired and checked by reading, not compiled, not run on hardware
//
#include <hip/hip_runtime.h>
#include <hip/hip_bf16.h>

// ---------------------------------------------------------------------------
// VQ-VAE forward for MI455X (gfx1250): all convs + VQ distance GEMM via
// v_wmma_f32_16x16x32_bf16 (bf16 in, f32 accumulate), b128 fragment loads.
// ---------------------------------------------------------------------------

typedef __attribute__((ext_vector_type(16))) __bf16 v16bf;
typedef __attribute__((ext_vector_type(8)))  float  v8f;

#define PIX   262144          // B*H*W = 16*128*128
#define NPRED 8388608         // 16*32*128*128

__device__ __forceinline__ unsigned short f2bf(float f) {
    union { float f; unsigned u; } x; x.f = f;
    unsigned r = x.u + 0x7FFFu + ((x.u >> 16) & 1u);   // round-nearest-even
    return (unsigned short)(r >> 16);
}
__device__ __forceinline__ float bf2f(unsigned short b) {
    union { unsigned u; float f; } x; x.u = ((unsigned)b) << 16;
    return x.f;
}

union Frag { uint4 q[2]; unsigned u[8]; v16bf v; };

// ---------------- layout conversion: images NCHW f32 -> NHWC bf16 ----------
__global__ __launch_bounds__(256)
void nchw2nhwc_bf16(const float* __restrict__ in, unsigned short* __restrict__ out, int C) {
    size_t i = (size_t)blockIdx.x * 256 + threadIdx.x;
    size_t total = (size_t)16 * C * 128 * 128;
    if (i >= total) return;
    int w = (int)(i & 127); size_t t = i >> 7;
    int h = (int)(t & 127); t >>= 7;
    int c = (int)(t % C);   int n = (int)(t / C);
    out[((size_t)((n * 128 + h) * 128 + w)) * C + c] = f2bf(in[i]);
}

// ---------------- weight repack: OIHW f32 -> B-fragment blocks -------------
// Layout: wrp[(((k>>5)*Cout + co)*2 + hl)*16 + kin], hl=(k>>4)&1, kin=k&15.
// A lane's B fragment (k = kk + 2v + 16*hl, 16 bf16) is then one contiguous
// 32-byte run -> two global_load_b128.
__global__ __launch_bounds__(256)
void repack_w(const float* __restrict__ w, unsigned short* __restrict__ wrp,
              int Cin, int Cout, int ksz) {
    int K = ksz * ksz * Cin;
    int total = K * Cout;
    int i = blockIdx.x * 256 + threadIdx.x;
    if (i >= total) return;
    int k  = i / Cout, co = i - k * Cout;
    int s  = k / Cin,  ci = k - s * Cin;
    int kh = s / ksz,  kw = s - kh * ksz;
    float v = w[(((size_t)co * Cin + ci) * ksz + kh) * ksz + kw];
    wrp[(((size_t)(k >> 5) * Cout + co) * 2 + ((k >> 4) & 1)) * 16 + (k & 15)] = f2bf(v);
}

// ---------------- codebook repack + ||E||^2 (same B-fragment layout) -------
__global__ __launch_bounds__(256)
void repack_codebook(const float* __restrict__ E, unsigned short* __restrict__ erp,
                     float* __restrict__ enorm) {
    int i = blockIdx.x * 256 + threadIdx.x;      // over 512*64
    if (i >= 512 * 64) return;
    int code = i >> 6, d = i & 63;
    erp[(((size_t)(d >> 5) * 512 + code) * 2 + ((d >> 4) & 1)) * 16 + (d & 15)] = f2bf(E[i]);
    if (d == 0) {
        float s = 0.f;
        for (int j = 0; j < 64; ++j) { float e = E[code * 64 + j]; s += e * e; }
        enorm[code] = s;
    }
}

// ---------------- implicit-GEMM conv, 16x16x32 bf16 WMMA -------------------
// block = 128 threads (4 waves); wave tile: 16 pixels x (16*NACC) couts.
// grid.x = PIX/64, grid.y = Cout/(16*NACC).
template <int NACC>
__global__ __launch_bounds__(128)
void conv_wmma(const unsigned short* __restrict__ in,   // NHWC bf16
               const unsigned short* __restrict__ wrp,  // packed B fragments
               const float* __restrict__ bias,
               const unsigned short* __restrict__ resid, // NHWC bf16 or null
               unsigned short* __restrict__ outb,        // NHWC bf16 or null
               float* __restrict__ outf,                 // NCHW f32 or null
               int Cin, int cinLog2, int Cout, int ksz, int relu) {
    const int lane = threadIdx.x & 31;
    const int wv   = threadIdx.x >> 5;
    const int hl   = lane >> 4;          // half-wave select
    const int l16  = lane & 15;
    const int mbase  = (blockIdx.x * 4 + wv) * 16;
    const int cobase = blockIdx.y * (16 * NACC);

    // this lane's A-row pixel
    const int pa = mbase + l16;
    const int wA = pa & 127, hA = (pa >> 7) & 127, nA = pa >> 14;

    const int pad  = ksz >> 1;
    const int Kdim = ksz * ksz * Cin;

    v8f acc[NACC];
#pragma unroll
    for (int n = 0; n < NACC; ++n) acc[n] = v8f{};

    for (int kk = 0; kk < Kdim; kk += 32) {
        const int sk = kk >> cinLog2;                 // kk / Cin
        const int cb = kk - (sk << cinLog2);
        const int kh = (ksz == 3) ? (sk / 3) : 0;
        const int kw = (ksz == 3) ? (sk - kh * 3) : 0;
        const int ih = hA + kh - pad, iw = wA + kw - pad;
        const bool ok = (ih >= 0) && (ih < 128) && (iw >= 0) && (iw < 128);

        // A fragment: lane covers k-local = 16*hl .. 16*hl+15: two b128
        Frag a;
        if (ok) {
            const unsigned short* ap =
                in + ((((size_t)((nA * 128 + ih) * 128 + iw)) << cinLog2) + cb) + hl * 16;
            a.q[0] = *(const uint4*)(ap);
            a.q[1] = *(const uint4*)(ap + 8);
        } else {
            a.q[0] = uint4{0u, 0u, 0u, 0u};
            a.q[1] = uint4{0u, 0u, 0u, 0u};
        }

        const size_t kt = (size_t)(kk >> 5) * Cout;
#pragma unroll
        for (int n = 0; n < NACC; ++n) {
            const unsigned short* bp =
                wrp + ((kt + cobase + n * 16 + l16) * 2 + hl) * 16;
            Frag b;
            b.q[0] = *(const uint4*)(bp);
            b.q[1] = *(const uint4*)(bp + 8);
            acc[n] = __builtin_amdgcn_wmma_f32_16x16x32_bf16(false, a.v, false, b.v,
                                                             (short)0, acc[n], false, false);
        }
    }

#pragma unroll
    for (int n = 0; n < NACC; ++n) {
        const int co = cobase + n * 16 + l16;
        const float bv = bias[co];
#pragma unroll
        for (int r = 0; r < 8; ++r) {
            const int p = mbase + r + hl * 8;      // C/D layout: M = r + 8*half
            float v = acc[n][r] + bv;
            if (resid) v += bf2f(resid[(size_t)p * Cout + co]);
            if (relu && v < 0.f) v = 0.f;
            if (outb) outb[(size_t)p * Cout + co] = f2bf(v);
            if (outf) {
                const int wo = p & 127, ho = (p >> 7) & 127, no = p >> 14;
                outf[(((size_t)no * Cout + co) * 128 + ho) * 128 + wo] = v;
            }
        }
    }
}

// ---------------- VQ argmin: -2*(z.E^T) + ||E||^2 via WMMA -----------------
__global__ __launch_bounds__(128)
void vq_argmin(const unsigned short* __restrict__ z,    // [PIX,64] bf16
               const unsigned short* __restrict__ erp,  // packed codebook
               const float* __restrict__ enorm,         // [512]
               int* __restrict__ idx) {
    const int lane = threadIdx.x & 31;
    const int wv   = threadIdx.x >> 5;
    const int hl   = lane >> 4, l16 = lane & 15;
    const int mbase = (blockIdx.x * 4 + wv) * 16;

    Frag a0, a1;
    const unsigned short* zp = z + (size_t)(mbase + l16) * 64 + hl * 16;
    a0.q[0] = *(const uint4*)(zp);
    a0.q[1] = *(const uint4*)(zp + 8);
    a1.q[0] = *(const uint4*)(zp + 32);
    a1.q[1] = *(const uint4*)(zp + 40);

    float minv[8]; int mini[8];
#pragma unroll
    for (int r = 0; r < 8; ++r) { minv[r] = 3.4e38f; mini[r] = 0; }

    for (int ct = 0; ct < 32; ++ct) {
        const int nb = ct * 16 + l16;
        const unsigned short* bp0 = erp + ((size_t)nb * 2 + hl) * 16;
        const unsigned short* bp1 = erp + (((size_t)512 + nb) * 2 + hl) * 16;
        if (ct < 31) {  // prefetch next code tile (global_prefetch_b8)
            __builtin_prefetch(bp0 + 32 * 16, 0, 0);
            __builtin_prefetch(bp1 + 32 * 16, 0, 0);
        }
        Frag b0, b1;
        b0.q[0] = *(const uint4*)(bp0);
        b0.q[1] = *(const uint4*)(bp0 + 8);
        b1.q[0] = *(const uint4*)(bp1);
        b1.q[1] = *(const uint4*)(bp1 + 8);
        v8f acc = {};
        acc = __builtin_amdgcn_wmma_f32_16x16x32_bf16(false, a0.v, false, b0.v,
                                                      (short)0, acc, false, false);
        acc = __builtin_amdgcn_wmma_f32_16x16x32_bf16(false, a1.v, false, b1.v,
                                                      (short)0, acc, false, false);
        const float en = enorm[nb];
#pragma unroll
        for (int r = 0; r < 8; ++r) {
            float d = en - 2.0f * acc[r];
            if (d < minv[r] || (d == minv[r] && nb < mini[r])) { minv[r] = d; mini[r] = nb; }
        }
    }
    // butterfly min-reduce across the 16 lanes of each half-group (wave32)
#pragma unroll
    for (int m = 8; m >= 1; m >>= 1) {
#pragma unroll
        for (int r = 0; r < 8; ++r) {
            float ov = __shfl_xor(minv[r], m, 32);
            int   oi = __shfl_xor(mini[r], m, 32);
            if (ov < minv[r] || (ov == minv[r] && oi < mini[r])) { minv[r] = ov; mini[r] = oi; }
        }
    }
    if (l16 == 0) {
#pragma unroll
        for (int r = 0; r < 8; ++r) idx[mbase + r + hl * 8] = mini[r];
    }
}

// ---------------- VQ losses: sum((z-zq)^2) and code counts -----------------
__global__ __launch_bounds__(256)
void vq_losses(const unsigned short* __restrict__ z, const float* __restrict__ E,
               const int* __restrict__ idx, float* __restrict__ emb_sum,
               float* __restrict__ counts) {
    int p = blockIdx.x * 256 + threadIdx.x;
    float s = 0.f;
    if (p < PIX) {
        int i = idx[p];
        const unsigned short* zp = z + (size_t)p * 64;
        const float* ep = E + (size_t)i * 64;
        for (int d = 0; d < 64; ++d) {
            float diff = bf2f(zp[d]) - ep[d];
            s += diff * diff;
        }
        atomicAdd(&counts[i], 1.0f);
    }
    __shared__ float red[256];
    red[threadIdx.x] = s;
    __syncthreads();
    for (int st = 128; st > 0; st >>= 1) {
        if (threadIdx.x < st) red[threadIdx.x] += red[threadIdx.x + st];
        __syncthreads();
    }
    if (threadIdx.x == 0) atomicAdd(emb_sum, red[0]);
}

// ---------------- reconstruction loss --------------------------------------
__global__ __launch_bounds__(256)
void recon_loss(const float* __restrict__ pred, const float* __restrict__ img,
                float* __restrict__ recon_sum, size_t n) {
    float s = 0.f;
    for (size_t i = (size_t)blockIdx.x * 256 + threadIdx.x; i < n;
         i += (size_t)gridDim.x * 256) {
        float d = pred[i] - img[i]; s += d * d;
    }
    __shared__ float red[256];
    red[threadIdx.x] = s;
    __syncthreads();
    for (int st = 128; st > 0; st >>= 1) {
        if (threadIdx.x < st) red[threadIdx.x] += red[threadIdx.x + st];
        __syncthreads();
    }
    if (threadIdx.x == 0) atomicAdd(recon_sum, red[0]);
}

// ---------------- finalize: loss + perplexity ------------------------------
__global__ __launch_bounds__(512)
void finalize_k(const float* __restrict__ sums, const float* __restrict__ counts,
                float* __restrict__ out_tail) {
    __shared__ float red[512];
    int t = threadIdx.x;
    float p = counts[t] / (float)PIX;
    red[t] = p * logf(p + 1e-10f);
    __syncthreads();
    for (int st = 256; st > 0; st >>= 1) {
        if (t < st) red[t] += red[t + st];
        __syncthreads();
    }
    if (t == 0) {
        float perplexity = expf(-red[0]);
        float emb   = 1.25f * (sums[0] / ((float)PIX * 64.0f)); // (1+beta)*mse
        float recon = sums[1] / (float)NPRED;
        out_tail[0] = recon + emb;
        out_tail[1] = perplexity;
    }
}

// ===========================================================================
extern "C" void kernel_launch(void* const* d_in, const int* in_sizes, int n_in,
                              void* d_out, int out_size, void* d_ws, size_t ws_size,
                              hipStream_t stream) {
    (void)in_sizes; (void)n_in; (void)out_size; (void)ws_size;
    const float* images = (const float*)d_in[0];
    const float* E      = (const float*)d_in[29];          // codebook [512,64]
    float* pred = (float*)d_out;                           // [16,32,128,128] NCHW

    char* ws = (char*)d_ws;
    float*          sums   = (float*)(ws + 0);             // [0]=emb, [1]=recon
    float*          counts = (float*)(ws + 256);           // [512]
    int*            idxb   = (int*)(ws + 4096);            // [PIX]
    unsigned short* wrp    = (unsigned short*)(ws + ((size_t)2 << 20));            // 512 KB
    unsigned short* erp    = (unsigned short*)(ws + ((size_t)2 << 20) + 0x80000);  // 64 KB
    float*          enorm  = (float*)(ws + ((size_t)2 << 20) + 0x90000);
    const size_t ab = ((size_t)3 << 20);
    unsigned short* A0 = (unsigned short*)(ws + ab);                          // 32 ch
    unsigned short* A1 = (unsigned short*)(ws + ab + ((size_t)16 << 20));     // 64 ch
    unsigned short* A2 = (unsigned short*)(ws + ab + ((size_t)48 << 20));     // 128 ch
    unsigned short* A3 = (unsigned short*)(ws + ab + ((size_t)112 << 20));    // 128 ch
    unsigned short* A4 = (unsigned short*)(ws + ab + ((size_t)176 << 20));    // 128 ch
    unsigned short* Zb = (unsigned short*)(ws + ab + ((size_t)240 << 20));    // 64 ch

    (void)hipMemsetAsync(d_ws, 0, 4096, stream);           // sums + counts

    auto conv = [&](const unsigned short* inb, int wi, int Cin, int cinLog2,
                    int Cout, int ksz, const unsigned short* resid,
                    unsigned short* outb, float* outf, int relu) {
        int total = ksz * ksz * Cin * Cout;
        repack_w<<<(total + 255) / 256, 256, 0, stream>>>(
            (const float*)d_in[wi], wrp, Cin, Cout, ksz);
        if (Cout >= 64) {
            dim3 g(PIX / 64, Cout / 64, 1);
            conv_wmma<4><<<g, 128, 0, stream>>>(inb, wrp, (const float*)d_in[wi + 1],
                                                resid, outb, outf,
                                                Cin, cinLog2, Cout, ksz, relu);
        } else {
            dim3 g(PIX / 64, Cout / 32, 1);
            conv_wmma<2><<<g, 128, 0, stream>>>(inb, wrp, (const float*)d_in[wi + 1],
                                                resid, outb, outf,
                                                Cin, cinLog2, Cout, ksz, relu);
        }
    };

    // ---- encoder ----
    nchw2nhwc_bf16<<<(NPRED + 255) / 256, 256, 0, stream>>>(images, A0, 32);
    conv(A0, 1, 32, 5, 64, 3, nullptr, A1, nullptr, 0);     // enc_in
    conv(A1, 3, 64, 6, 128, 3, nullptr, A2, nullptr, 1);    // enc_rb1.c1 + relu
    conv(A1, 7, 64, 6, 128, 1, nullptr, A4, nullptr, 0);    // enc_rb1.skip
    conv(A2, 5, 128, 7, 128, 3, A4, A3, nullptr, 1);        // enc_rb1.c2 + skip + relu
    conv(A3, 9, 128, 7, 128, 3, nullptr, A2, nullptr, 1);   // enc_rb2.c1 + relu
    conv(A2, 11, 128, 7, 128, 3, A3, A4, nullptr, 1);       // enc_rb2.c2 + x + relu
    conv(A4, 13, 128, 7, 64, 3, nullptr, Zb, nullptr, 0);   // enc_out -> z (relu(x)==x)

    // ---- quantize ----
    repack_codebook<<<(512 * 64 + 255) / 256, 256, 0, stream>>>(E, erp, enorm);
    vq_argmin<<<PIX / 64, 128, 0, stream>>>(Zb, erp, enorm, idxb);
    vq_losses<<<PIX / 256, 256, 0, stream>>>(Zb, E, idxb, &sums[0], counts);

    // ---- decoder (consumes raw z, per reference) ----
    conv(Zb, 15, 64, 6, 128, 3, nullptr, A2, nullptr, 0);   // dec_in
    conv(A2, 17, 128, 7, 128, 3, nullptr, A3, nullptr, 1);  // dec_rb1.c1 + relu
    conv(A3, 19, 128, 7, 128, 3, A2, A4, nullptr, 1);       // dec_rb1.c2 + x + relu
    conv(A4, 25, 128, 7, 64, 1, nullptr, A1, nullptr, 0);   // dec_rb2.skip
    conv(A4, 21, 128, 7, 64, 3, nullptr, A3, nullptr, 1);   // dec_rb2.c1 + relu
    conv(A3, 23, 64, 6, 64, 3, A1, A2, nullptr, 1);         // dec_rb2.c2 + skip + relu
    conv(A2, 27, 64, 6, 32, 3, nullptr, nullptr, pred, 0);  // dec_out -> f32 NCHW

    // ---- losses ----
    recon_loss<<<4096, 256, 0, stream>>>(pred, images, &sums[1], (size_t)NPRED);
    finalize_k<<<1, 512, 0, stream>>>(sums, counts, pred + NPRED);
}
